// Gate_65060164600304
// MI455X (gfx1250) — matrix operand order repaired and checked by reading, not compiled, not run on hardware
//
#include <hip/hip_runtime.h>

typedef __attribute__((ext_vector_type(16))) __bf16 v16bf;
typedef __attribute__((ext_vector_type(8)))  __bf16 v8bf;
typedef __attribute__((ext_vector_type(4)))  __bf16 v4bf;
typedef __attribute__((ext_vector_type(8)))  float  v8f;

#define T_TOKENS 16384
#define DIM      2048
#define NEXP     256
#define NGROUP   8
#define EPG      32
#define TOPK     8
#define NLIM     4
#define RSCALE   2.5f

#define XS_LD (DIM + 8)    // bf16 stride: 4112 B/row -> 16B-aligned rows, 4-bank skew
#define LS_LD (NEXP + 4)   // float stride for score rows

__global__ void zero_load_kernel(float* __restrict__ p) {
    p[threadIdx.x] = 0.0f;
}

// One-shot weight cast f32 -> bf16 into workspace (2 MB -> 1 MB, L2-resident)
__global__ __launch_bounds__(256) void wcast_kernel(const float* __restrict__ w,
                                                    __bf16* __restrict__ wbf) {
    const size_t i4 = ((size_t)blockIdx.x * 256 + threadIdx.x) * 4;
    const float4 v = *(const float4*)(w + i4);
    v4bf o;
    o[0] = (__bf16)v.x; o[1] = (__bf16)v.y; o[2] = (__bf16)v.z; o[3] = (__bf16)v.w;
    *(v4bf*)(wbf + i4) = o;
}

__global__ __launch_bounds__(256) void gate_fused_kernel(
    const float*  __restrict__ x,      // [T, D] f32
    const __bf16* __restrict__ wbf,    // [E, D] bf16 (pre-cast)
    const float*  __restrict__ bias,   // [E]
    float* __restrict__ wts_out,       // [T, 8]
    float* __restrict__ idx_out,       // [T, 8] (indices stored as float)
    float* __restrict__ load_out)      // [E]
{
    __shared__ __bf16 xs[16 * XS_LD];        // 16x2048 x-tile in bf16
    __shared__ float  scores_s[16 * LS_LD];  // raw logits
    __shared__ float  bias_s[NEXP];

    const int tid  = threadIdx.x;
    const int tok0 = blockIdx.x * 16;

    // ---- Phase 1: stage x tile to LDS as bf16 (float4 loads, 8B bf16 stores) ----
    for (int i = tid * 4; i < 16 * DIM; i += 256 * 4) {
        const int r = i >> 11;            // DIM = 2048
        const int c = i & (DIM - 1);
        const float4 v = *(const float4*)(x + (size_t)(tok0 + r) * DIM + c);
        v4bf o;
        o[0] = (__bf16)v.x; o[1] = (__bf16)v.y; o[2] = (__bf16)v.z; o[3] = (__bf16)v.w;
        *(v4bf*)(xs + r * XS_LD + c) = o;
    }
    if (tid < NEXP) bias_s[tid] = bias[tid];
    __syncthreads();

    // ---- Phase 2: WMMA GEMM: 16 tokens x 256 experts, K = 2048 ----
    const int lane = tid & 31;
    const int wv   = tid >> 5;            // wave 0..7 -> experts [wv*32, wv*32+32)
    const int lhi  = lane >> 4;           // 0 | 1  (half-wave)
    const int l15  = lane & 15;
    const int n0   = wv * 32;

    v8f acc0 = {};
    v8f acc1 = {};

    const __bf16* __restrict__ xrow  = &xs[l15 * XS_LD];
    const __bf16* __restrict__ wrow0 = &wbf[(size_t)(n0 + l15) * DIM];
    const __bf16* __restrict__ wrow1 = &wbf[(size_t)(n0 + 16 + l15) * DIM];

    for (int k0 = 0; k0 < DIM; k0 += 32) {
        // A fragment (16x32 bf16): half j -> K = k0 + lhi*16 + j
        const v8bf alo = *(const v8bf*)(xrow + k0 + (lhi << 4));
        const v8bf ahi = *(const v8bf*)(xrow + k0 + 8 + (lhi << 4));
        const v16bf a = __builtin_shufflevector(alo, ahi,
                            0, 1, 2, 3, 4, 5, 6, 7, 8, 9, 10, 11, 12, 13, 14, 15);
        // B fragments (32x16 bf16): half h -> K = k0 + lhi*16 + h, col N = n0(+16)+l15
        const int kb = k0 + (lhi << 4);
        const v16bf b0 = *(const v16bf*)(wrow0 + kb);
        const v16bf b1 = *(const v16bf*)(wrow1 + kb);

        __builtin_prefetch(wrow0 + kb + 32, 0, 0);
        __builtin_prefetch(wrow1 + kb + 32, 0, 0);

        acc0 = __builtin_amdgcn_wmma_f32_16x16x32_bf16(
                   false, a, false, b0, (short)0, acc0, false, false);
        acc1 = __builtin_amdgcn_wmma_f32_16x16x32_bf16(
                   false, a, false, b1, (short)0, acc1, false, false);
    }

    // ---- Phase 3: spill C fragments to LDS (lane=N, VGPR v -> M = v + 8*lhi) ----
    {
        const int m = 8 * lhi;
#pragma unroll
        for (int v = 0; v < 8; ++v) {
            scores_s[(v + m) * LS_LD + n0 + l15]      = acc0[v];
            scores_s[(v + m) * LS_LD + n0 + 16 + l15] = acc1[v];
        }
    }
    __syncthreads();

    // ---- Phase 4: routing, 16 lanes per token (2 tokens per wave) ----
    {
        const int t    = 2 * wv + lhi;    // token 0..15 within tile
        const int base = l15 * 16;        // this lane's 16-expert chunk
        const float* __restrict__ row = &scores_s[t * LS_LD];

        float sc[16], selv[16];

        // local logits + max
        float mx = -__builtin_inff();
#pragma unroll
        for (int j = 0; j < 16; ++j) {
            sc[j] = row[base + j];
            mx = fmaxf(mx, sc[j]);
        }
#pragma unroll
        for (int off = 1; off < 16; off <<= 1)
            mx = fmaxf(mx, __shfl_xor(mx, off, 32));

        // exp + sum
        float ssum = 0.0f;
#pragma unroll
        for (int j = 0; j < 16; ++j) {
            sc[j] = __expf(sc[j] - mx);
            ssum += sc[j];
        }
#pragma unroll
        for (int off = 1; off < 16; off <<= 1)
            ssum += __shfl_xor(ssum, off, 32);
        const float inv = 1.0f / ssum;

        // scores (for gather) and selection scores (bias-adjusted)
#pragma unroll
        for (int j = 0; j < 16; ++j) {
            sc[j]   = sc[j] * inv;
            selv[j] = sc[j] + bias_s[base + j];
        }

        // group scores: chunk = half a group; shfl_xor(1) completes the group max
        float gm = -__builtin_inff();
#pragma unroll
        for (int j = 0; j < 16; ++j) gm = fmaxf(gm, selv[j]);
        gm = fmaxf(gm, __shfl_xor(gm, 1, 32));
        // broadcast all 8 group maxes to every lane of this token
        float gs[NGROUP];
#pragma unroll
        for (int g = 0; g < NGROUP; ++g)
            gs[g] = __shfl(gm, (lhi << 4) + (g << 1), 32);

        // top-4 groups, computed redundantly on every lane (strict '>' = low-idx tiebreak)
        unsigned gmask = 0u;
#pragma unroll
        for (int i = 0; i < NLIM; ++i) {
            float best = -__builtin_inff(); int bg = 0;
#pragma unroll
            for (int g = 0; g < NGROUP; ++g) {
                const bool used = (gmask >> g) & 1u;
                if (!used && gs[g] > best) { best = gs[g]; bg = g; }
            }
            gmask |= (1u << bg);
        }
        // mask: this lane's chunk belongs to group l15>>1
        if (!((gmask >> (l15 >> 1)) & 1u)) {
#pragma unroll
            for (int j = 0; j < 16; ++j) selv[j] = -__builtin_inff();
        }

        // top-8 experts: local argmax (regs) + 4-stage shuffle reduce, 8 passes
        const size_t tg = (size_t)(tok0 + t);
#pragma unroll
        for (int p = 0; p < TOPK; ++p) {
            float bv = -__builtin_inff(), bs = 0.0f; int be = 0;
#pragma unroll
            for (int j = 0; j < 16; ++j) {
                if (selv[j] > bv) { bv = selv[j]; bs = sc[j]; be = base + j; }
            }
#pragma unroll
            for (int off = 1; off < 16; off <<= 1) {
                const float ov = __shfl_xor(bv, off, 32);
                const float os = __shfl_xor(bs, off, 32);
                const int   oe = __shfl_xor(be, off, 32);
                if (ov > bv || (ov == bv && oe < be)) { bv = ov; bs = os; be = oe; }
            }
            // clear the winner in its owning lane (compare-based: stays in registers)
#pragma unroll
            for (int j = 0; j < 16; ++j) {
                if (base + j == be) selv[j] = -__builtin_inff();
            }
            if (l15 == 0) {
                wts_out[tg * TOPK + p] = bs * RSCALE;
                idx_out[tg * TOPK + p] = (float)be;
                atomicAdd(&load_out[be], 1.0f);
            }
        }
    }
}

extern "C" void kernel_launch(void* const* d_in, const int* in_sizes, int n_in,
                              void* d_out, int out_size, void* d_ws, size_t ws_size,
                              hipStream_t stream) {
    const float* x    = (const float*)d_in[0];   // [16384, 2048] f32
    const float* w    = (const float*)d_in[1];   // [256, 2048]  f32
    const float* bias = (const float*)d_in[2];   // [256]        f32

    float* out  = (float*)d_out;
    float* wts  = out;                                   // [T*8]
    float* idx  = out + (size_t)T_TOKENS * TOPK;         // [T*8]
    float* load = out + 2 * (size_t)T_TOKENS * TOPK;     // [256]

    __bf16* wbf = (__bf16*)d_ws;                         // [256*2048] bf16 (1 MB)

    zero_load_kernel<<<1, NEXP, 0, stream>>>(load);
    wcast_kernel<<<(NEXP * DIM) / (256 * 4), 256, 0, stream>>>(w, wbf);
    gate_fused_kernel<<<T_TOKENS / 16, 256, 0, stream>>>(x, wbf, bias, wts, idx, load);
}
